// Experts_62852551410286
// MI455X (gfx1250) — matrix operand order repaired and checked by reading, not compiled, not run on hardware
//
#include <hip/hip_runtime.h>

// ---------------------------------------------------------------------------
// Grouped expert MLP  (E=8, B=2, N=2048, D=1024, F=4096), M = B*N = 4096
//   out[e] = (gelu_tanh(X[e] @ W1[e] + b1[e])) @ W2[e] + b2[e]
// Per expert:
//   1) convert X_e  f32 -> bf16            (8 MB)
//   2) transpose+convert W1_e -> W1t[F][D] (8 MB), W2_e -> W2t[D][F] (8 MB)
//   3) GEMM1: H = gelu(Xb @ W1) -> bf16 H[M][F] in ws
//   4) GEMM2: out = H @ W2 + b2 -> f32
// GEMMs: 128x128x32 tiles, 8 wave32s, v_wmma_f32_16x16x32_bf16, double-
// buffered LDS fed by global_load_async_to_lds_b128 (ASYNCcnt engine).
// ---------------------------------------------------------------------------

typedef __bf16 v16bf __attribute__((ext_vector_type(16)));
typedef float  v8f   __attribute__((ext_vector_type(8)));

#define BM 128
#define BN 128
#define BK 32
#define LDA 40            // BK + 8 pad (bf16) -> 80B row stride (16B aligned)
#define LDB 40
#define THREADS 256       // 8 wave32s

union Frag { uint4 u[2]; v16bf v; };

__device__ __forceinline__ float gelu_tanh(float x) {
  const float k0 = 0.7978845608028654f;   // sqrt(2/pi)
  const float k1 = 0.044715f;
  float t = tanhf(k0 * (x + k1 * x * x * x));
  return 0.5f * x * (1.0f + t);
}

// Low 32 bits of a generic pointer to LDS == LDS byte offset (aperture rule).
__device__ __forceinline__ uint32_t lds_u32(const void* p) {
  return (uint32_t)(uintptr_t)p;
}

// One per-lane 16B async copy global -> LDS (tracked by ASYNCcnt).
__device__ __forceinline__ void async_cp_b128(uint32_t lds_byte, const void* g) {
  asm volatile("global_load_async_to_lds_b128 %0, %1, off"
               :: "v"(lds_byte), "v"((uint64_t)(uintptr_t)g) : "memory");
}
__device__ __forceinline__ void wait_async0() {
  asm volatile("s_wait_asynccnt 0x0" ::: "memory");
}

// ---------------- elementwise f32 -> bf16 (8 elems/thread) -----------------
__global__ __launch_bounds__(256)
void cvt_bf16x8(const float* __restrict__ src, __bf16* __restrict__ dst, int n8) {
  const int i = blockIdx.x * blockDim.x + threadIdx.x;
  if (i >= n8) return;
  const float4 a = ((const float4*)src)[2 * i + 0];
  const float4 b = ((const float4*)src)[2 * i + 1];
  union { __bf16 h[8]; uint4 u; } p;
  p.h[0] = (__bf16)a.x; p.h[1] = (__bf16)a.y; p.h[2] = (__bf16)a.z; p.h[3] = (__bf16)a.w;
  p.h[4] = (__bf16)b.x; p.h[5] = (__bf16)b.y; p.h[6] = (__bf16)b.z; p.h[7] = (__bf16)b.w;
  ((uint4*)dst)[i] = p.u;
}

// ------------- tiled transpose + convert: src[R][C] -> dst[C][R] -----------
__global__ __launch_bounds__(256)
void transpose_cvt(const float* __restrict__ src, __bf16* __restrict__ dst,
                   int R, int C) {
  __shared__ float t[32][33];
  const int x = blockIdx.x * 32 + threadIdx.x;
  const int y0 = blockIdx.y * 32;
  #pragma unroll
  for (int j = threadIdx.y; j < 32; j += 8)
    t[j][threadIdx.x] = src[(size_t)(y0 + j) * C + x];
  __syncthreads();
  const int xo  = blockIdx.y * 32 + threadIdx.x;
  const int yo0 = blockIdx.x * 32;
  #pragma unroll
  for (int j = threadIdx.y; j < 32; j += 8)
    dst[(size_t)(yo0 + j) * R + xo] = (__bf16)t[threadIdx.x][j];
}

// ---------------------------- bf16 WMMA GEMM -------------------------------
// C = epilogue(A(MxK) @ B + bias), B given transposed: Bt[N][K]. All bf16 in.
template <bool GELU, typename OT>
__global__ __launch_bounds__(THREADS)
void gemm_bf16(const __bf16* __restrict__ A, const __bf16* __restrict__ Bt,
               const float* __restrict__ bias, OT* __restrict__ Cout,
               int M, int K, int Ncol)
{
  __shared__ __align__(16) __bf16 As[2][BM][LDA];
  __shared__ __align__(16) __bf16 Bs[2][BN][LDB];

  const int tid  = threadIdx.x;
  const int lane = tid & 31;
  const int wave = tid >> 5;        // 0..7
  const int wm   = wave >> 1;       // 0..3 : 32-row strip
  const int wn   = wave & 1;        // 0..1 : 64-col strip
  const int half = lane >> 4;       // 0/1
  const int lr   = lane & 15;       // 0..15

  const int mb = blockIdx.y * BM;
  const int nb = blockIdx.x * BN;

  // ---- async staging slots: each thread copies 2x16B for A and for B
  const int sr = tid >> 2;                 // 0..63 : row within block
  const int sk = (tid & 3) * 8;            // bf16 k-offset (0/8/16/24)
  const __bf16* ga0 = A  + (size_t)(mb + sr)      * K + sk;
  const __bf16* ga1 = A  + (size_t)(mb + sr + 64) * K + sk;
  const __bf16* gb0 = Bt + (size_t)(nb + sr)      * K + sk;
  const __bf16* gb1 = Bt + (size_t)(nb + sr + 64) * K + sk;
  const uint32_t aBuf = BM * LDA * 2;      // bytes per A buffer
  const uint32_t bBuf = BN * LDB * 2;
  const uint32_t la = lds_u32(&As[0][sr][sk]);
  const uint32_t lb = lds_u32(&Bs[0][sr][sk]);
  const uint32_t aRow64 = 64 * LDA * 2;
  const uint32_t bRow64 = 64 * LDB * 2;

  auto stage = [&](int buf, int kb) {
    async_cp_b128(la + buf * aBuf,          ga0 + kb);
    async_cp_b128(la + buf * aBuf + aRow64, ga1 + kb);
    async_cp_b128(lb + buf * bBuf,          gb0 + kb);
    async_cp_b128(lb + buf * bBuf + bRow64, gb1 + kb);
  };

  v8f acc[2][4];
  #pragma unroll
  for (int i = 0; i < 2; ++i)
    #pragma unroll
    for (int j = 0; j < 4; ++j) acc[i][j] = {};

  stage(0, 0);
  wait_async0();
  __syncthreads();

  int cur = 0;
  for (int kb = 0; kb < K; kb += BK) {
    const int nxt = cur ^ 1;
    if (kb + BK < K) stage(nxt, kb + BK);   // overlap copy with WMMA below

    v16bf afrag[2], bfrag[4];
    #pragma unroll
    for (int tm = 0; tm < 2; ++tm) {        // A 16x32: M=lr, K chunks
      const __bf16* s = &As[cur][wm * 32 + tm * 16 + lr][0];
      Frag f;
      f.u[0] = *(const uint4*)(s + half * 8);        // K = half*8 .. +7
      f.u[1] = *(const uint4*)(s + 16 + half * 8);   // K = 16+half*8 .. +7
      afrag[tm] = f.v;
    }
    #pragma unroll
    for (int tn = 0; tn < 4; ++tn) {        // B 32x16: N=lr, K chunks half*8, 16+half*8
      const __bf16* s = &Bs[cur][wn * 64 + tn * 16 + lr][0];
      Frag f;
      f.u[0] = *(const uint4*)(s + half * 8);
      f.u[1] = *(const uint4*)(s + 16 + half * 8);
      bfrag[tn] = f.v;
    }
    #pragma unroll
    for (int tm = 0; tm < 2; ++tm)
      #pragma unroll
      for (int tn = 0; tn < 4; ++tn)
        acc[tm][tn] = __builtin_amdgcn_wmma_f32_16x16x32_bf16(
            false, afrag[tm], false, bfrag[tn], (short)0, acc[tm][tn],
            false, false);

    wait_async0();            // next buffer fully written (per-wave) ...
    __syncthreads();          // ... and visible to all waves
    cur = nxt;
  }

  // ---- epilogue: C/D layout => element i: row = half*8 + i, col = lr
  #pragma unroll
  for (int tm = 0; tm < 2; ++tm) {
    #pragma unroll
    for (int tn = 0; tn < 4; ++tn) {
      const int col = nb + wn * 64 + tn * 16 + lr;
      const float bv = bias[col];
      #pragma unroll
      for (int i = 0; i < 8; ++i) {
        const int row = mb + wm * 32 + tm * 16 + half * 8 + i;
        float v = acc[tm][tn][i] + bv;
        if constexpr (GELU) v = gelu_tanh(v);
        Cout[(size_t)row * Ncol + col] = (OT)v;
      }
    }
  }
}

extern "C" void kernel_launch(void* const* d_in, const int* in_sizes, int n_in,
                              void* d_out, int out_size, void* d_ws, size_t ws_size,
                              hipStream_t stream) {
  (void)in_sizes; (void)n_in; (void)out_size; (void)ws_size;
  const float* x  = (const float*)d_in[0];   // (E,B,N,D) flat == (E, M, D)
  const float* w1 = (const float*)d_in[1];   // (E, D, F)
  const float* b1 = (const float*)d_in[2];   // (E, F)
  const float* w2 = (const float*)d_in[3];   // (E, F, D)
  const float* b2 = (const float*)d_in[4];   // (E, D)
  float* out = (float*)d_out;                // (E, M, D) flat

  const int E = 8, D = 1024, F = 4096, M = 4096;

  // workspace layout (bf16 elements): 56 MB total
  __bf16* Xb  = (__bf16*)d_ws;               // M x D
  __bf16* W1t = Xb  + (size_t)M * D;         // F x D  (= W1^T)
  __bf16* W2t = W1t + (size_t)D * F;         // D x F  (= W2^T)
  __bf16* H   = W2t + (size_t)F * D;         // M x F

  dim3 blk(THREADS);
  dim3 tblk(32, 8);
  dim3 g1(F / BN, M / BM);                   // 32 x 32
  dim3 g2(D / BN, M / BM);                   //  8 x 32

  for (int e = 0; e < E; ++e) {
    cvt_bf16x8<<<(M * D / 8 + 255) / 256, 256, 0, stream>>>(
        x + (size_t)e * M * D, Xb, M * D / 8);
    transpose_cvt<<<dim3(F / 32, D / 32), tblk, 0, stream>>>(
        w1 + (size_t)e * D * F, W1t, D, F);          // [D][F] -> [F][D]
    transpose_cvt<<<dim3(D / 32, F / 32), tblk, 0, stream>>>(
        w2 + (size_t)e * F * D, W2t, F, D);          // [F][D] -> [D][F]
    gemm_bf16<true, __bf16><<<g1, blk, 0, stream>>>(
        Xb, W1t, b1 + (size_t)e * F, H, M, D, F);
    gemm_bf16<false, float><<<g2, blk, 0, stream>>>(
        H, W2t, b2 + (size_t)e * D, out + (size_t)e * M * D, M, F, D);
  }
}
